// EdgeRegressionNetWithGAT_71768903516436
// MI455X (gfx1250) — matrix operand order repaired and checked by reading, not compiled, not run on hardware
//
#include <hip/hip_runtime.h>
#include <hip/hip_bf16.h>

#define HEADS 8
#define HID 32
#define CH 256            // HEADS*HID
#define NEG_SLOPE 0.2f

typedef __attribute__((ext_vector_type(16))) __bf16 v16bf;
typedef __attribute__((ext_vector_type(8)))  float  v8f;

// ---------------------------------------------------------------------------
// WMMA fragment helpers (layouts per CDNA5 ISA 7.12.2, wave32)
// ---------------------------------------------------------------------------

// A fragment: 16x32 bf16, rows m0..m0+15 (optionally gathered through idx),
// K window [kc, kc+32). Per lane this is two contiguous 8-element runs ->
// compiler emits 2x global_load_b128.
__device__ __forceinline__ v16bf load_a_frag(const __bf16* A, int lda,
                                             const int* idx, int m0, int kc,
                                             int lane) {
    int m = lane & 15, half = lane >> 4;
    int r = idx ? idx[m0 + m] : (m0 + m);
    const __bf16* p = A + (size_t)r * lda + kc + half * 8;
    v16bf a;
#pragma unroll
    for (int v = 0; v < 8; ++v) {
        int kb = ((v & 3) * 2) + ((v & 4) ? 16 : 0);
        a[2 * v]     = p[kb];
        a[2 * v + 1] = p[kb + 1];
    }
    return a;
}

// Packed-B pointer: weights are pre-packed fragment-major so each lane's 16
// bf16 values are contiguous (32 B). One 32-byte vector load per fragment.
__device__ __forceinline__ v16bf load_b_packed(const __bf16* Bp, int kt,
                                               int nt, int NT, int lane) {
    return *(const v16bf*)(Bp + (((size_t)kt * NT + nt) * 32 + lane) * 16);
}

// ---------------------------------------------------------------------------
// Pack f32 row-major [K x N] weights into WMMA B-fragment-major bf16.
// Fragment layout: lane = {half, n}; element j -> k = kt*32 + half*8 + (j&7) + (j>>3)*16,
// n = nt*16 + (lane & 15). Destination index == flat thread id.
// ---------------------------------------------------------------------------
__global__ void k_pack_b(const float* __restrict__ W, __bf16* __restrict__ Bp,
                         int K, int N) {
    int t = blockIdx.x * blockDim.x + threadIdx.x;
    if (t >= K * N) return;
    int j    = t & 15;
    int lane = (t >> 4) & 31;
    int rest = t >> 9;
    int NT   = N >> 4;
    int nt   = rest % NT;
    int kt   = rest / NT;
    int k = kt * 32 + (lane >> 4) * 8 + (j & 7) + (j >> 3) * 16;
    int n = nt * 16 + (lane & 15);
    Bp[t] = (__bf16)W[(size_t)k * N + n];
}

// ---------------------------------------------------------------------------
// Generic GEMM: C[M,N] = act(A[M,K] * Bp[K,N] + bias). Each wave computes a
// 16x32 C tile (one A fragment reused for two WMMAs). M mult 16, N mult 32,
// K mult 32.
// ---------------------------------------------------------------------------
__global__ __launch_bounds__(128) void gemm_wmma(
    const __bf16* __restrict__ A, const __bf16* __restrict__ Bp,
    const float* __restrict__ bias, float* __restrict__ Cf,
    __bf16* __restrict__ Cb, int M, int N, int K, int do_relu) {
    int lane = threadIdx.x & 31;
    int wave = threadIdx.x >> 5;
    int m0  = (blockIdx.x * 4 + wave) * 16;
    int nt0 = blockIdx.y * 2;
    int NT  = N >> 4;
    if (m0 >= M) return;  // wave-uniform

    v8f c0 = {}, c1 = {};
    int KT = K >> 5;
    for (int kt = 0; kt < KT; ++kt) {
        v16bf a  = load_a_frag(A, K, nullptr, m0, kt * 32, lane);
        v16bf b0 = load_b_packed(Bp, kt, nt0, NT, lane);
        v16bf b1 = load_b_packed(Bp, kt, nt0 + 1, NT, lane);
        c0 = __builtin_amdgcn_wmma_f32_16x16x32_bf16(false, a, false, b0,
                                                     (short)0, c0, false, false);
        c1 = __builtin_amdgcn_wmma_f32_16x16x32_bf16(false, a, false, b1,
                                                     (short)0, c1, false, false);
    }
    int n = lane & 15, half = lane >> 4;
    auto store_tile = [&](v8f& cc, int nt) {
        float bv = bias ? bias[nt * 16 + n] : 0.f;
#pragma unroll
        for (int r = 0; r < 8; ++r) {
            float v = cc[r] + bv;
            if (do_relu) v = v > 0.f ? v : 0.f;
            size_t o = (size_t)(m0 + r + half * 8) * N + nt * 16 + n;
            if (Cf) Cf[o] = v;
            if (Cb) Cb[o] = (__bf16)v;
        }
    };
    store_tile(c0, nt0);
    store_tile(c1, nt0 + 1);
}

// ---------------------------------------------------------------------------
// Fused edge predictor GEMM: for each edge e,
//   Out[e,:] = relu( x2[row[e]] @ Wp[0:256] + x2[col[e]] @ Wp[256:512]
//                    + emb[e] @ Wp[512:544] + bias )
// The [E,544] concat matrix is never materialized; gathered A fragments are
// each reused for both output n-tiles (N = 32).
// ---------------------------------------------------------------------------
__global__ __launch_bounds__(128) void pred_gemm_wmma(
    const __bf16* __restrict__ Xb, const __bf16* __restrict__ Eb,
    const __bf16* __restrict__ Wp, const float* __restrict__ bias,
    const int* __restrict__ row, const int* __restrict__ col,
    float* __restrict__ Out, int E) {
    int lane = threadIdx.x & 31;
    int wave = threadIdx.x >> 5;
    int e0 = (blockIdx.x * 4 + wave) * 16;
    if (e0 >= E) return;

    const int NT = 2;  // N = 32
    v8f c0 = {}, c1 = {};
#pragma unroll 1
    for (int kt = 0; kt < 8; ++kt) {
        v16bf a  = load_a_frag(Xb, CH, row, e0, kt * 32, lane);
        v16bf b0 = load_b_packed(Wp, kt, 0, NT, lane);
        v16bf b1 = load_b_packed(Wp, kt, 1, NT, lane);
        c0 = __builtin_amdgcn_wmma_f32_16x16x32_bf16(false, a, false, b0,
                                                     (short)0, c0, false, false);
        c1 = __builtin_amdgcn_wmma_f32_16x16x32_bf16(false, a, false, b1,
                                                     (short)0, c1, false, false);
        a  = load_a_frag(Xb, CH, col, e0, kt * 32, lane);
        b0 = load_b_packed(Wp, 8 + kt, 0, NT, lane);
        b1 = load_b_packed(Wp, 8 + kt, 1, NT, lane);
        c0 = __builtin_amdgcn_wmma_f32_16x16x32_bf16(false, a, false, b0,
                                                     (short)0, c0, false, false);
        c1 = __builtin_amdgcn_wmma_f32_16x16x32_bf16(false, a, false, b1,
                                                     (short)0, c1, false, false);
    }
    {   // edge-embedding block, K = 32 (kt = 16)
        v16bf a  = load_a_frag(Eb, HID, nullptr, e0, 0, lane);
        v16bf b0 = load_b_packed(Wp, 16, 0, NT, lane);
        v16bf b1 = load_b_packed(Wp, 16, 1, NT, lane);
        c0 = __builtin_amdgcn_wmma_f32_16x16x32_bf16(false, a, false, b0,
                                                     (short)0, c0, false, false);
        c1 = __builtin_amdgcn_wmma_f32_16x16x32_bf16(false, a, false, b1,
                                                     (short)0, c1, false, false);
    }
    int n = lane & 15, half = lane >> 4;
    auto store_tile = [&](v8f& cc, int nt) {
        float bv = bias[nt * 16 + n];
#pragma unroll
        for (int r = 0; r < 8; ++r) {
            float v = cc[r] + bv;
            v = v > 0.f ? v : 0.f;
            Out[(size_t)(e0 + r + half * 8) * HID + nt * 16 + n] = v;
        }
    };
    store_tile(c0, 0);
    store_tile(c1, 1);
}

// ---------------------------------------------------------------------------
// Elementwise / attention kernels
// ---------------------------------------------------------------------------
__global__ void k_fill_f32(float* p, float v, size_t n) {
    size_t i = (size_t)blockIdx.x * blockDim.x + threadIdx.x;
    if (i < n) p[i] = v;
}

__global__ void k_f32_to_bf16(const float* __restrict__ s,
                              __bf16* __restrict__ d, size_t n) {
    size_t i = (size_t)blockIdx.x * blockDim.x + threadIdx.x;
    if (i < n) d[i] = (__bf16)s[i];
}

// alpha_src[n,h] = dot(h[n,h*32: ], a_src[h]); same for dst
__global__ void k_alpha(const float* __restrict__ h,
                        const float* __restrict__ a_src,
                        const float* __restrict__ a_dst,
                        float* __restrict__ asrc, float* __restrict__ adst,
                        int n_nodes) {
    int i = blockIdx.x * blockDim.x + threadIdx.x;
    if (i >= n_nodes * HEADS) return;
    int node = i / HEADS, hd = i % HEADS;
    const float* hp = h + (size_t)node * CH + hd * HID;
    const float* sp = a_src + hd * HID;
    const float* dp = a_dst + hd * HID;
    float s = 0.f, d = 0.f;
#pragma unroll
    for (int j = 0; j < HID; ++j) { s += hp[j] * sp[j]; d += hp[j] * dp[j]; }
    asrc[i] = s;
    adst[i] = d;
}

__device__ __forceinline__ float atomicMaxFloat(float* addr, float val) {
    return (val >= 0.f)
        ? __int_as_float(atomicMax((int*)addr, __float_as_int(val)))
        : __uint_as_float(atomicMin((unsigned int*)addr, __float_as_uint(val)));
}

// e = leaky_relu(asrc[src]+adst[dst]); segment max into m[dst]
__global__ void k_attn_logits(const int* __restrict__ row,
                              const int* __restrict__ col,
                              const float* __restrict__ asrc,
                              const float* __restrict__ adst,
                              float* __restrict__ e, float* __restrict__ m,
                              int E, int n_nodes) {
    int idx = blockIdx.x * blockDim.x + threadIdx.x;
    int tot = (E + n_nodes) * HEADS;
    if (idx >= tot) return;
    int i = idx / HEADS, hd = idx % HEADS;
    int src = (i < E) ? row[i] : (i - E);
    int dst = (i < E) ? col[i] : (i - E);
    float v = asrc[src * HEADS + hd] + adst[dst * HEADS + hd];
    v = (v > 0.f) ? v : NEG_SLOPE * v;
    e[idx] = v;
    atomicMaxFloat(&m[dst * HEADS + hd], v);
}

// ex = exp(e - m[dst]) (in place); segment sum into denom[dst]
__global__ void k_attn_exp(const int* __restrict__ col,
                           float* __restrict__ e, const float* __restrict__ m,
                           float* __restrict__ denom, int E, int n_nodes) {
    int idx = blockIdx.x * blockDim.x + threadIdx.x;
    int tot = (E + n_nodes) * HEADS;
    if (idx >= tot) return;
    int i = idx / HEADS, hd = idx % HEADS;
    int dst = (i < E) ? col[i] : (i - E);
    float ex = __expf(e[idx] - m[dst * HEADS + hd]);
    e[idx] = ex;
    atomicAdd(&denom[dst * HEADS + hd], ex);
}

// agg[dst,c] += h[src,c] * ex/denom[dst]; one block per edge, 256 ch threads
__global__ __launch_bounds__(256) void k_attn_scatter(
    const int* __restrict__ row, const int* __restrict__ col,
    const float* __restrict__ h, const float* __restrict__ ex,
    const float* __restrict__ denom, float* __restrict__ agg, int E) {
    int i = blockIdx.x;
    int t = threadIdx.x;           // channel 0..255
    int hd = t >> 5;
    int src = (i < E) ? row[i] : (i - E);
    int dst = (i < E) ? col[i] : (i - E);
    float alpha = ex[(size_t)i * HEADS + hd] / denom[dst * HEADS + hd];
    atomicAdd(&agg[(size_t)dst * CH + t], h[(size_t)src * CH + t] * alpha);
}

// x_next = bf16(relu(agg + b))
__global__ void k_relu_bias_bf16(const float* __restrict__ agg,
                                 const float* __restrict__ b,
                                 __bf16* __restrict__ xb, size_t total) {
    size_t i = (size_t)blockIdx.x * blockDim.x + threadIdx.x;
    if (i >= total) return;
    float v = agg[i] + b[i % CH];
    xb[i] = (__bf16)(v > 0.f ? v : 0.f);
}

// out[e] = dot(predh[e,:32], Wp2) + bp2
__global__ void k_final_dot(const float* __restrict__ ph,
                            const float* __restrict__ Wp2,
                            const float* __restrict__ bp2,
                            float* __restrict__ out, int E) {
    int i = blockIdx.x * blockDim.x + threadIdx.x;
    if (i >= E) return;
    const float* p = ph + (size_t)i * HID;
    float s = 0.f;
#pragma unroll
    for (int j = 0; j < HID; ++j) s += p[j] * Wp2[j];
    out[i] = s + bp2[0];
}

// ---------------------------------------------------------------------------
// Host launcher
// ---------------------------------------------------------------------------
extern "C" void kernel_launch(void* const* d_in, const int* in_sizes, int n_in,
                              void* d_out, int out_size, void* d_ws,
                              size_t ws_size, hipStream_t stream) {
    const float* x    = (const float*)d_in[0];
    const int*   ei   = (const int*)d_in[1];
    const float* ea   = (const float*)d_in[2];
    const float* W1   = (const float*)d_in[3];
    const float* as1  = (const float*)d_in[4];
    const float* ad1  = (const float*)d_in[5];
    const float* b1   = (const float*)d_in[6];
    const float* W2   = (const float*)d_in[7];
    const float* as2  = (const float*)d_in[8];
    const float* ad2  = (const float*)d_in[9];
    const float* b2   = (const float*)d_in[10];
    const float* Wm1  = (const float*)d_in[11];
    const float* bm1  = (const float*)d_in[12];
    const float* Wm2  = (const float*)d_in[13];
    const float* bm2  = (const float*)d_in[14];
    const float* Wp1  = (const float*)d_in[15];
    const float* bp1  = (const float*)d_in[16];
    const float* Wp2  = (const float*)d_in[17];
    const float* bp2  = (const float*)d_in[18];

    const int NODE_IN = 128;
    const int Nn = in_sizes[0] / NODE_IN;      // 50000
    const int E  = in_sizes[1] / 2;            // 400000
    const int* row = ei;
    const int* col = ei + E;

    // workspace carve-up (256B aligned)
    char* wp = (char*)d_ws;
    auto alloc = [&](size_t bytes) -> void* {
        void* p = wp;
        wp += (bytes + 255) & ~(size_t)255;
        return p;
    };
    __bf16* xb    = (__bf16*)alloc((size_t)Nn * CH * 2);   // node feats bf16
    float*  h     = (float*)alloc((size_t)Nn * CH * 4);
    float*  agg   = (float*)alloc((size_t)Nn * CH * 4);
    float*  asrc  = (float*)alloc((size_t)Nn * HEADS * 4);
    float*  adst  = (float*)alloc((size_t)Nn * HEADS * 4);
    float*  mmax  = (float*)alloc((size_t)Nn * HEADS * 4);
    float*  den   = (float*)alloc((size_t)Nn * HEADS * 4);
    float*  ebuf  = (float*)alloc((size_t)(E + Nn) * HEADS * 4);
    __bf16* W1b   = (__bf16*)alloc((size_t)NODE_IN * CH * 2);   // packed
    __bf16* W2b   = (__bf16*)alloc((size_t)CH * CH * 2);        // packed
    __bf16* Wm1b  = (__bf16*)alloc((size_t)HID * HID * 2);      // packed
    __bf16* Wm2b  = (__bf16*)alloc((size_t)HID * HID * 2);      // packed
    __bf16* Wp1b  = (__bf16*)alloc((size_t)(2 * CH + HID) * HID * 2); // packed
    __bf16* eb0   = (__bf16*)alloc((size_t)E * HID * 2);
    __bf16* eb1   = (__bf16*)alloc((size_t)E * HID * 2);
    __bf16* embb  = (__bf16*)alloc((size_t)E * HID * 2);
    float*  predh = (float*)alloc((size_t)E * HID * 4);

    auto blk = [](size_t n, int b) { return (unsigned)((n + b - 1) / b); };

    // weight packing (fragment-major bf16)
    k_pack_b<<<blk((size_t)NODE_IN * CH, 256), 256, 0, stream>>>(W1, W1b, NODE_IN, CH);
    k_pack_b<<<blk((size_t)CH * CH, 256), 256, 0, stream>>>(W2, W2b, CH, CH);
    k_pack_b<<<blk((size_t)HID * HID, 256), 256, 0, stream>>>(Wm1, Wm1b, HID, HID);
    k_pack_b<<<blk((size_t)HID * HID, 256), 256, 0, stream>>>(Wm2, Wm2b, HID, HID);
    k_pack_b<<<blk((size_t)(2 * CH + HID) * HID, 256), 256, 0, stream>>>(Wp1, Wp1b, 2 * CH + HID, HID);
    // input features -> bf16 (layer-1 input, lda = 128)
    k_f32_to_bf16<<<blk((size_t)Nn * NODE_IN, 256), 256, 0, stream>>>(x, xb, (size_t)Nn * NODE_IN);

    int mtiles_n = (Nn + 15) / 16;
    int mtiles_e = (E + 15) / 16;
    int attn_tot = (E + Nn) * HEADS;

    auto run_gat = [&](int K, const __bf16* Wb, const float* a_s,
                       const float* a_d, const float* bias) {
        // h = x @ W  (f32 accumulate)
        gemm_wmma<<<dim3(blk(mtiles_n, 4), CH / 32), 128, 0, stream>>>(
            xb, Wb, nullptr, h, nullptr, Nn, CH, K, 0);
        // per-node attention logits
        k_alpha<<<blk((size_t)Nn * HEADS, 256), 256, 0, stream>>>(h, a_s, a_d, asrc, adst, Nn);
        // init segment buffers
        k_fill_f32<<<blk((size_t)Nn * HEADS, 256), 256, 0, stream>>>(mmax, -3.0e38f, (size_t)Nn * HEADS);
        k_fill_f32<<<blk((size_t)Nn * HEADS, 256), 256, 0, stream>>>(den, 0.f, (size_t)Nn * HEADS);
        k_fill_f32<<<blk((size_t)Nn * CH, 256), 256, 0, stream>>>(agg, 0.f, (size_t)Nn * CH);
        // segment softmax (self-loops make every segment non-empty)
        k_attn_logits<<<blk(attn_tot, 256), 256, 0, stream>>>(row, col, asrc, adst, ebuf, mmax, E, Nn);
        k_attn_exp<<<blk(attn_tot, 256), 256, 0, stream>>>(col, ebuf, mmax, den, E, Nn);
        // weighted message scatter
        k_attn_scatter<<<E + Nn, 256, 0, stream>>>(row, col, h, ebuf, den, agg, E);
        // x_next = bf16(relu(agg + b))
        k_relu_bias_bf16<<<blk((size_t)Nn * CH, 256), 256, 0, stream>>>(agg, bias, xb, (size_t)Nn * CH);
    };

    run_gat(NODE_IN, W1b, as1, ad1, b1);  // layer 1: K=128
    run_gat(CH, W2b, as2, ad2, b2);       // layer 2: K=256; xb now holds x2

    // edge MLP: relu(relu(ea@Wm1+bm1)@Wm2+bm2) in bf16
    k_f32_to_bf16<<<blk((size_t)E * HID, 256), 256, 0, stream>>>(ea, eb0, (size_t)E * HID);
    gemm_wmma<<<dim3(blk(mtiles_e, 4), 1), 128, 0, stream>>>(
        eb0, Wm1b, bm1, nullptr, eb1, E, HID, HID, 1);
    gemm_wmma<<<dim3(blk(mtiles_e, 4), 1), 128, 0, stream>>>(
        eb1, Wm2b, bm2, nullptr, embb, E, HID, HID, 1);

    // fused gather + predictor GEMM (K = 544 split 256+256+32)
    pred_gemm_wmma<<<blk(mtiles_e, 4), 128, 0, stream>>>(
        xb, embb, Wp1b, bp1, row, col, predh, E);

    // final projection to scalar per edge
    k_final_dot<<<blk(E, 256), 256, 0, stream>>>(predh, Wp2, bp2, (float*)d_out, E);
}
